// type3_nufft_81415400063468
// MI455X (gfx1250) — matrix operand order repaired and checked by reading, not compiled, not run on hardware
//
#include <hip/hip_runtime.h>

// y[n,t] = sum_r x[n,r] * exp(-2i*pi * sum_b phis[b,r]*alphas[b,t])
// B=3, R=32768, T=4096, N=4.  Fused phase -> sincos -> WMMA matmul.
//
// Per block: one 16-wide t tile. Per wave (8 waves): one R/8 slice, reduced in
// chunks of K=32 with v_wmma_f32_16x16x32_f16 (M=16 rows, channels padded 4->16).
// sin/cos use the raw v_sin_f32/v_cos_f32 semantics (argument in revolutions:
// sin(2*pi*x)), so the phase dot-product feeds the trans ops with ZERO extra
// multiplies: the -1 sign is folded into pre-negated alphas.
// Deterministic LDS reduction across the 8 waves, then write complex64 output.

typedef __attribute__((ext_vector_type(16))) _Float16 v16h;
typedef __attribute__((ext_vector_type(8)))  float    v8f;
typedef __attribute__((ext_vector_type(4)))  float    v4f;

__global__ __launch_bounds__(256) void type3_nufft_wmma_kernel(
    const float* __restrict__ phis,    // (3, R)
    const float* __restrict__ alphas,  // (3, T)
    const float* __restrict__ x,       // (4, R)
    float* __restrict__ out,           // (4, T) complex64 -> interleaved re,im
    int R, int T)
{
  const int tid   = threadIdx.x;
  const int lane  = tid & 31;
  const int w     = tid >> 5;            // wave index 0..7
  const int tbase = blockIdx.x << 4;     // 16 t-values per block

  // ---- B-fragment mapping (16-bit B matrix, 32x16 KxN, contiguous K order used here) ----
  // lane = (k/16)*16 + n ; element e of v16h = k%16
  const int nCol = lane & 15;            // t column within tile
  const int kB   = (lane >> 4) << 4;     // K offset: 0 or 16
  const int t    = tbase + nCol;

  // ---- A-fragment mapping (16-bit A matrix, 16x32 MxK, contiguous K order used here) ----
  // lane<16: e=0..15 -> K=0..15 ; lane>=16: K+=16 (matches B's contiguous K)
  const int mRow  = lane & 15;           // channel row (only 0..3 real)
  const int koffA = (lane >> 4) ? 16 : 0;
  const float xscale = (mRow < 4) ? 1.0f : 0.0f;   // branchless zero-pad rows 4..15
  const float* __restrict__ xr_base = x + (size_t)(mRow & 3) * R;

  // per-wave r slice
  const int rSlice = R >> 3;             // 4096
  const int rBegin = w * rSlice;
  const int nChunk = rSlice >> 5;        // 128 chunks of 32

  // NEGATED alphas for this lane's t column: the phase dot-product then
  // directly produces -phase, which v_sin/v_cos consume as revolutions.
  const float nal0 = -alphas[0 * T + t];
  const float nal1 = -alphas[1 * T + t];
  const float nal2 = -alphas[2 * T + t];

  const float* __restrict__ ph0 = phis;
  const float* __restrict__ ph1 = phis + R;
  const float* __restrict__ ph2 = phis + 2 * R;

  v8f accRe = {};
  v8f accIm = {};

  for (int c = 0; c < nChunk; ++c) {
    const int rbase = rBegin + (c << 5);
    const int r0 = rbase + kB;           // 16 consecutive r for this lane's B frag

    // prefetch next chunk's phis (gfx1250 global_prefetch_b8 path)
    if (c + 1 < nChunk) {
      __builtin_prefetch(ph0 + r0 + 32, 0, 3);
      __builtin_prefetch(ph1 + r0 + 32, 0, 3);
      __builtin_prefetch(ph2 + r0 + 32, 0, 3);
    }

    // ---- build B fragments ----
    // q = -phase;  exp(-2i*pi*phase) = cos(2*pi*q) + i*sin(2*pi*q)
    // v_cos_f32(q) = cos(2*pi*q), v_sin_f32(q) = sin(2*pi*q): no arg scaling.
    v16h bCos, bSin;
#pragma unroll
    for (int g = 0; g < 4; ++g) {
      const int rg = r0 + 4 * g;
      v4f p0 = *reinterpret_cast<const v4f*>(ph0 + rg);
      v4f p1 = *reinterpret_cast<const v4f*>(ph1 + rg);
      v4f p2 = *reinterpret_cast<const v4f*>(ph2 + rg);
#pragma unroll
      for (int j = 0; j < 4; ++j) {
        float q = fmaf(p0[j], nal0, fmaf(p1[j], nal1, p2[j] * nal2)); // -phase
        float s = __builtin_amdgcn_sinf(q);   // v_sin_f32: sin(2*pi*q)
        float cc = __builtin_amdgcn_cosf(q);  // v_cos_f32: cos(2*pi*q)
        bCos[4 * g + j] = (_Float16)cc;
        bSin[4 * g + j] = (_Float16)s;
      }
    }

    // ---- build A fragment: x rows (channels), rows 4..15 zeroed ----
    const float* __restrict__ xr = xr_base + rbase;
    v4f x0 = *reinterpret_cast<const v4f*>(xr + koffA);        // K = koffA+0..3
    v4f x1 = *reinterpret_cast<const v4f*>(xr + koffA + 4);    // K = koffA+4..7
    v4f x2 = *reinterpret_cast<const v4f*>(xr + koffA + 8);    // K = koffA+8..11
    v4f x3 = *reinterpret_cast<const v4f*>(xr + koffA + 12);   // K = koffA+12..15
    v16h a;
#pragma unroll
    for (int j = 0; j < 4; ++j) {
      a[j]      = (_Float16)(x0[j] * xscale);
      a[4 + j]  = (_Float16)(x1[j] * xscale);
      a[8 + j]  = (_Float16)(x2[j] * xscale);
      a[12 + j] = (_Float16)(x3[j] * xscale);
    }

    // ---- D = A*B + C : two WMMAs (Re with cos, Im with sin) ----
    accRe = __builtin_amdgcn_wmma_f32_16x16x32_f16(
        false, a, false, bCos, (short)0, accRe, false, false);
    accIm = __builtin_amdgcn_wmma_f32_16x16x32_f16(
        false, a, false, bSin, (short)0, accIm, false, false);
  }

  // ---- deterministic cross-wave reduction via LDS ----
  // C/D layout: lane l, VGPR v -> row M = v + 8*(l>=16), col N = l&15.
  __shared__ float red[8][2][8][32];
#pragma unroll
  for (int v = 0; v < 8; ++v) {
    red[w][0][v][lane] = accRe[v];
    red[w][1][v][lane] = accIm[v];
  }
  __syncthreads();

  // 128 useful outputs per block: 2 parts x 4 channels x 16 t
  if (tid < 128) {
    const int tloc = tid & 15;
    const int mm   = (tid >> 4) & 3;     // channel 0..3 -> VGPR row mm, lanes 0..15
    const int p    = tid >> 6;           // 0 = re, 1 = im
    float s = 0.0f;
#pragma unroll
    for (int ww = 0; ww < 8; ++ww) s += red[ww][p][mm][tloc];
    out[((size_t)mm * T + tbase + tloc) * 2 + p] = s;
  }
}

extern "C" void kernel_launch(void* const* d_in, const int* in_sizes, int n_in,
                              void* d_out, int out_size, void* d_ws, size_t ws_size,
                              hipStream_t stream) {
  const float* phis   = (const float*)d_in[0];   // (B=3, R)
  const float* alphas = (const float*)d_in[1];   // (B=3, T)
  const float* x      = (const float*)d_in[2];   // (N=4, R)
  float* out = (float*)d_out;                    // (N, T) complex64 interleaved

  const int B = 3;
  const int R = in_sizes[0] / B;                 // 32768
  const int T = in_sizes[1] / B;                 // 4096

  dim3 grid(T / 16);
  dim3 block(256);
  hipLaunchKernelGGL(type3_nufft_wmma_kernel, grid, block, 0, stream,
                     phis, alphas, x, out, R, T);
}
